// CausalGraphAttn_26250840113640
// MI455X (gfx1250) — matrix operand order repaired and checked by reading, not compiled, not run on hardware
//
#include <hip/hip_runtime.h>
#include <hip/hip_bf16.h>

// ---------------------------------------------------------------------------
// CausalGraphAttn for MI455X (gfx1250, wave32, WMMA f16 16x16x32, f32 accum)
// GEMMs use double-buffered GLOBAL_LOAD_ASYNC_TO_LDS_B128 (ASYNCcnt) staging.
// ---------------------------------------------------------------------------

typedef __attribute__((ext_vector_type(16))) _Float16 v16h;
typedef __attribute__((ext_vector_type(8)))  _Float16 v8h;
typedef __attribute__((ext_vector_type(8)))  float    v8f;

#define D_MODEL 1024
#define N_SEQ   2048
#define BATCH   4
#define N_HEADS 16
#define HEAD_DIM 64
#define ROWS    (BATCH * N_SEQ)   // 8192

__device__ __forceinline__ v8f wmma32(v16h a, v16h b, v8f c) {
  return __builtin_amdgcn_wmma_f32_16x16x32_f16(
      /*neg_a=*/false, a, /*neg_b=*/false, b,
      /*c_mod=*/(short)0, c, /*reuse_a=*/false, /*reuse_b=*/false);
}

// A fragment (16x32 f16, M=rows): lane L holds row L&15.
// halves 0..15 = K (L>=16 ? 16 : 0) + 0..15 (matches load_bfrag)
__device__ __forceinline__ v16h load_afrag(const _Float16* base, int ld) {
  int lane = threadIdx.x & 31;
  int row  = lane & 15;
  int grp  = lane >> 4;
  const _Float16* p = base + (size_t)row * ld + grp * 16;
  union { v16h v; v8h h[2]; } u;
  u.h[0] = *(const v8h*)(p);
  u.h[1] = *(const v8h*)(p + 8);
  return u.v;
}

// B fragment (32x16 f16, K x N) stored as rows of N with K contiguous:
// lane L holds column L&15; halves 0..15 = K (L>=16 ? 16 : 0) + 0..15
__device__ __forceinline__ v16h load_bfrag(const _Float16* base, int ld) {
  int lane = threadIdx.x & 31;
  int col  = lane & 15;
  int grp  = lane >> 4;
  return *(const v16h*)(base + (size_t)col * ld + grp * 16);
}

// Async 16-byte copy global -> LDS (per-lane), tracked by ASYNCcnt.
// LDS aperture keeps the group-relative offset in the low 32 address bits.
__device__ __forceinline__ void async_copy16(void* lds, const _Float16* g) {
  unsigned lds_off = (unsigned)(uintptr_t)lds;
  unsigned long long ga = (unsigned long long)(uintptr_t)g;
  asm volatile("global_load_async_to_lds_b128 %0, %1, off"
               :: "v"(lds_off), "v"(ga) : "memory");
}

// ---------------------------------------------------------------------------
// Kernel 1: LayerNorm (f32 in) -> f16 out, one block per row of 1024
// ---------------------------------------------------------------------------
__global__ __launch_bounds__(256) void ln_kernel(
    const float* __restrict__ X, const float* __restrict__ gamma,
    const float* __restrict__ beta, _Float16* __restrict__ H)
{
  __shared__ float red[256];
  int row = blockIdx.x;
  int t = threadIdx.x;
  const float* x = X + (size_t)row * D_MODEL;
  float v[4];
  float s = 0.f;
#pragma unroll
  for (int i = 0; i < 4; ++i) { v[i] = x[t + 256 * i]; s += v[i]; }
  red[t] = s; __syncthreads();
  for (int off = 128; off > 0; off >>= 1) {
    if (t < off) red[t] += red[t + off];
    __syncthreads();
  }
  float mu = red[0] * (1.f / D_MODEL);
  __syncthreads();
  float s2 = 0.f;
#pragma unroll
  for (int i = 0; i < 4; ++i) { float d = v[i] - mu; s2 += d * d; }
  red[t] = s2; __syncthreads();
  for (int off = 128; off > 0; off >>= 1) {
    if (t < off) red[t] += red[t + off];
    __syncthreads();
  }
  float rstd = rsqrtf(red[0] * (1.f / D_MODEL) + 1e-5f);
  _Float16* o = H + (size_t)row * D_MODEL;
#pragma unroll
  for (int i = 0; i < 4; ++i) {
    int c = t + 256 * i;
    o[c] = (_Float16)((v[i] - mu) * rstd * gamma[c] + beta[c]);
  }
}

// ---------------------------------------------------------------------------
// Kernel 2: f32 -> f16 cast (weights)
// ---------------------------------------------------------------------------
__global__ __launch_bounds__(256) void cast_f16_kernel(
    const float* __restrict__ in, _Float16* __restrict__ out, int n)
{
  int i = blockIdx.x * 256 + threadIdx.x;
  if (i < n) out[i] = (_Float16)in[i];
}

// ---------------------------------------------------------------------------
// Kernel 3: pack no_edge[n][m] = (adj[m][n] < 0.1) into bitmask words
// mb[n*64 + m/32], bit (m&31)
// ---------------------------------------------------------------------------
__global__ __launch_bounds__(256) void maskpack_kernel(
    const float* __restrict__ adj, unsigned* __restrict__ mb)
{
  int idx = blockIdx.x * 256 + threadIdx.x;   // 2048*64 words
  int n = idx & (N_SEQ - 1);
  int w = idx >> 11;
  unsigned bits = 0u;
#pragma unroll
  for (int bit = 0; bit < 32; ++bit) {
    int m = w * 32 + bit;
    if (adj[(size_t)m * N_SEQ + n] < 0.1f) bits |= (1u << bit);
  }
  mb[(size_t)n * 64 + w] = bits;
}

// ---------------------------------------------------------------------------
// GEMM tiling shared by QKV and out-projection:
//   block tile 64(M) x 256(N), K-step 32, 8 waves each computing 32x64
//   (2x4 WMMA fragments = 8 wmma per K-step), double-buffered async staging.
// Per thread per K-tile: 1 A-chunk + 4 B-chunks = 5 async b128 copies.
// ---------------------------------------------------------------------------
#define GEMM_TN 256

// Issue one K-tile's async copies into LDS buffer (5 per thread).
__device__ __forceinline__ void gemm_issue_tile(
    const _Float16* __restrict__ Asrc, const _Float16* __restrict__ Bsrc,
    _Float16* sA, _Float16* sB, int m0, int n0, int kk, int t)
{
  {
    int r = t >> 2, c = (t & 3) * 8;
    async_copy16(&sA[r * 32 + c], &Asrc[(size_t)(m0 + r) * D_MODEL + kk + c]);
  }
#pragma unroll
  for (int i = 0; i < 4; ++i) {
    int id = t + i * 256;
    int r = id >> 2, c = (id & 3) * 8;
    async_copy16(&sB[r * 32 + c], &Bsrc[(size_t)(n0 + r) * D_MODEL + kk + c]);
  }
}

__device__ __forceinline__ void gemm_core(
    const _Float16* __restrict__ Asrc, const _Float16* __restrict__ Bsrc,
    _Float16 (*sA)[64 * 32], _Float16 (*sB)[GEMM_TN * 32],
    int m0, int n0, int t, v8f acc[2][4])
{
  int wave = t >> 5;
  int wm = wave >> 2;          // 0..1 : 32-row sub-tile
  int wn = wave & 3;           // 0..3 : 64-col sub-tile

  gemm_issue_tile(Asrc, Bsrc, sA[0], sB[0], m0, n0, 0, t);
  int iter = 0;
  for (int kk = 0; kk < D_MODEL; kk += 32, ++iter) {
    int cur = iter & 1;
    bool more = (kk + 32 < D_MODEL);
    if (more)
      gemm_issue_tile(Asrc, Bsrc, sA[cur ^ 1], sB[cur ^ 1], m0, n0, kk + 32, t);
    // Async loads complete in order: waiting to <=5 outstanding means the
    // current tile's 5 copies have landed; the next tile's 5 stay in flight.
    if (more) asm volatile("s_wait_asynccnt 0x5" ::: "memory");
    else      asm volatile("s_wait_asynccnt 0x0" ::: "memory");
    __syncthreads();

    const _Float16* A0 = sA[cur];
    const _Float16* B0 = sB[cur];
    v16h a0 = load_afrag(&A0[(wm * 32 + 0)  * 32], 32);
    v16h a1 = load_afrag(&A0[(wm * 32 + 16) * 32], 32);
    v16h b[4];
#pragma unroll
    for (int j = 0; j < 4; ++j)
      b[j] = load_bfrag(&B0[(wn * 64 + j * 16) * 32], 32);
#pragma unroll
    for (int j = 0; j < 4; ++j) {
      acc[0][j] = wmma32(a0, b[j], acc[0][j]);
      acc[1][j] = wmma32(a1, b[j], acc[1][j]);
    }
    __syncthreads();   // all waves done reading before buffer is refilled
  }
}

// ---------------------------------------------------------------------------
// Kernel 4: QKV GEMM  qkv = H(f16)[8192,1024] @ Wqkv(f16)[3072,1024]^T
// Epilogue scatters into per-head Q[b,h,n,d], K[b,h,n,d], V^T[b,h,d,n] (f16).
// ---------------------------------------------------------------------------
__global__ __launch_bounds__(256) void gemm_qkv_kernel(
    const _Float16* __restrict__ H, const _Float16* __restrict__ W,
    _Float16* __restrict__ Q, _Float16* __restrict__ K,
    _Float16* __restrict__ VT)
{
  __shared__ __align__(32) _Float16 sA[2][64 * 32];
  __shared__ __align__(32) _Float16 sB[2][GEMM_TN * 32];
  int m0 = blockIdx.x * 64;
  int n0 = blockIdx.y * GEMM_TN;
  int t = threadIdx.x;
  int wave = t >> 5;
  int wm = wave >> 2;
  int wn = wave & 3;
  v8f acc[2][4] = {};

  gemm_core(H, W, sA, sB, m0, n0, t, acc);

  // Epilogue: C fragment element (row = r + 8*(lane>>4), col = lane&15)
  int lane = t & 31;
  int rowoff = (lane >> 4) * 8;
  int coloff = lane & 15;
#pragma unroll
  for (int i = 0; i < 2; ++i)
#pragma unroll
    for (int j = 0; j < 4; ++j)
#pragma unroll
      for (int r = 0; r < 8; ++r) {
        int gm = m0 + wm * 32 + i * 16 + rowoff + r;   // [0, 8192)
        int gn = n0 + wn * 64 + j * 16 + coloff;       // [0, 3072)
        float v = acc[i][j][r];
        int b = gm >> 11;
        int n = gm & (N_SEQ - 1);
        int which = gn >> 10;                 // 0=q 1=k 2=v
        int h = (gn & 1023) >> 6;
        int d = gn & 63;
        size_t bh = (size_t)b * N_HEADS + h;
        if (which == 0)
          Q[bh * N_SEQ * HEAD_DIM + (size_t)n * HEAD_DIM + d] = (_Float16)v;
        else if (which == 1)
          K[bh * N_SEQ * HEAD_DIM + (size_t)n * HEAD_DIM + d] = (_Float16)v;
        else
          VT[bh * HEAD_DIM * N_SEQ + (size_t)d * N_SEQ + n] = (_Float16)v;
      }
}

// ---------------------------------------------------------------------------
// Kernel 5: flash attention. Grid = B*H*(N/64); 4 waves, 16 query rows each.
// S = (Q/8) K^T via WMMA, online softmax in C-fragment layout, P via LDS
// re-layout to an A fragment, O += P @ V^T via WMMA.
// ---------------------------------------------------------------------------
__global__ __launch_bounds__(128) void attn_kernel(
    const _Float16* __restrict__ Q, const _Float16* __restrict__ K,
    const _Float16* __restrict__ VT, const unsigned* __restrict__ mb,
    _Float16* __restrict__ O)
{
  __shared__ __align__(32) _Float16 sP[4][16 * 64];

  int bid = blockIdx.x;
  int rt  = bid & 31;            // N/64 row tiles
  int bh  = bid >> 5;            // b*16 + h
  int h   = bh & (N_HEADS - 1);
  int b   = bh >> 4;
  int t = threadIdx.x;
  int wave = t >> 5;
  int lane = t & 31;
  int hi  = lane >> 4;
  int col = lane & 15;
  int row0 = rt * 64 + wave * 16;
  const bool masked_head = (h < 4);

  const _Float16* Qh = Q  + (size_t)bh * N_SEQ * HEAD_DIM;
  const _Float16* Kh = K  + (size_t)bh * N_SEQ * HEAD_DIM;
  const _Float16* Vh = VT + (size_t)bh * HEAD_DIM * N_SEQ;

  // Q fragments (scaled by 1/sqrt(64) = 0.125)
  v16h qf[2];
#pragma unroll
  for (int ks = 0; ks < 2; ++ks) {
    v16h a = load_afrag(Qh + (size_t)row0 * HEAD_DIM + ks * 32, HEAD_DIM);
#pragma unroll
    for (int e = 0; e < 16; ++e) a[e] = a[e] * (_Float16)0.125f;
    qf[ks] = a;
  }

  float mstat[8], lstat[8];
#pragma unroll
  for (int r = 0; r < 8; ++r) { mstat[r] = -1e30f; lstat[r] = 0.f; }
  v8f oacc[4] = {};
  _Float16* P = sP[wave];

  for (int mc = 0; mc < N_SEQ; mc += 64) {
    // ---- S tile 16x64 (4 column fragments), K over hd=64 in 2 WMMA steps
    v8f s[4];
#pragma unroll
    for (int f = 0; f < 4; ++f) {
      v8f c = {};
#pragma unroll
      for (int ks = 0; ks < 2; ++ks) {
        v16h bf = load_bfrag(Kh + (size_t)(mc + f * 16) * HEAD_DIM + ks * 32,
                             HEAD_DIM);
        c = wmma32(qf[ks], bf, c);
      }
      s[f] = c;
    }
    // ---- graph mask on first N_CAUSAL heads
    if (masked_head) {
#pragma unroll
      for (int r = 0; r < 8; ++r) {
        int n = row0 + r + 8 * hi;
        unsigned w0 = mb[(size_t)n * 64 + (mc >> 5)];
        unsigned w1 = mb[(size_t)n * 64 + (mc >> 5) + 1];
#pragma unroll
        for (int f = 0; f < 4; ++f) {
          int c2 = col + 16 * f;
          unsigned w = (c2 < 32) ? w0 : w1;
          if ((w >> (c2 & 31)) & 1u) s[f][r] = -1e9f;
        }
      }
    }
    // ---- online softmax: row max / rescale / exp / row sum
    float newm[8], alpha[8];
#pragma unroll
    for (int r = 0; r < 8; ++r) {
      float mx = fmaxf(fmaxf(s[0][r], s[1][r]), fmaxf(s[2][r], s[3][r]));
#pragma unroll
      for (int d = 1; d < 16; d <<= 1)
        mx = fmaxf(mx, __shfl_xor(mx, d, 32));
      float nm = fmaxf(mstat[r], mx);
      alpha[r] = __expf(mstat[r] - nm);
      newm[r] = nm;
    }
#pragma unroll
    for (int r = 0; r < 8; ++r) {
      float sum = 0.f;
#pragma unroll
      for (int f = 0; f < 4; ++f) {
        float p = __expf(s[f][r] - newm[r]);
        s[f][r] = p;
        sum += p;
      }
#pragma unroll
      for (int d = 1; d < 16; d <<= 1)
        sum += __shfl_xor(sum, d, 32);
      lstat[r] = lstat[r] * alpha[r] + sum;
      mstat[r] = newm[r];
    }
#pragma unroll
    for (int f = 0; f < 4; ++f)
#pragma unroll
      for (int r = 0; r < 8; ++r)
        oacc[f][r] *= alpha[r];
    // ---- C-layout -> row-major P tile in LDS (wave-local)
#pragma unroll
    for (int f = 0; f < 4; ++f)
#pragma unroll
      for (int r = 0; r < 8; ++r)
        P[(r + 8 * hi) * 64 + col + 16 * f] = (_Float16)s[f][r];
    asm volatile("s_wait_dscnt 0" ::: "memory");   // LDS RAW, same wave
    // ---- O += P @ V^T  (4 output d-fragments x 2 K steps)
#pragma unroll
    for (int f = 0; f < 4; ++f) {
#pragma unroll
      for (int ks = 0; ks < 2; ++ks) {
        v16h pa = load_afrag(P + ks * 32, 64);
        v16h vb = load_bfrag(Vh + (size_t)(f * 16) * N_SEQ + mc + ks * 32,
                             N_SEQ);
        oacc[f] = wmma32(pa, vb, oacc[f]);
      }
    }
    asm volatile("s_wait_dscnt 0" ::: "memory");   // finish reads before next P
  }

  // ---- normalize and store attn_out[b][n][h*64 + d] as f16
  _Float16* Ob = O + (size_t)b * N_SEQ * D_MODEL;
#pragma unroll
  for (int r = 0; r < 8; ++r) {
    float inv = 1.0f / fmaxf(lstat[r], 1e-30f);
#pragma unroll
    for (int f = 0; f < 4; ++f) {
      int n = row0 + r + 8 * hi;
      int d = f * 16 + col;
      Ob[(size_t)n * D_MODEL + h * HEAD_DIM + d] = (_Float16)(oacc[f][r] * inv);
    }
  }
}

// ---------------------------------------------------------------------------
// Kernel 6: out projection  y = AttnOut(f16) @ Wout(f16)^T + b_out + x  (f32)
// ---------------------------------------------------------------------------
__global__ __launch_bounds__(256) void gemm_out_kernel(
    const _Float16* __restrict__ A, const _Float16* __restrict__ W,
    const float* __restrict__ bias, const float* __restrict__ X,
    float* __restrict__ Y)
{
  __shared__ __align__(32) _Float16 sA[2][64 * 32];
  __shared__ __align__(32) _Float16 sB[2][GEMM_TN * 32];
  int m0 = blockIdx.x * 64;
  int n0 = blockIdx.y * GEMM_TN;
  int t = threadIdx.x;
  int wave = t >> 5;
  int wm = wave >> 2;
  int wn = wave & 3;
  v8f acc[2][4] = {};

  gemm_core(A, W, sA, sB, m0, n0, t, acc);

  int lane = t & 31;
  int rowoff = (lane >> 4) * 8;
  int coloff = lane & 15;
#pragma unroll
  for (int i = 0; i < 2; ++i)
#pragma unroll
    for (int j = 0; j < 4; ++j)
#pragma unroll
      for (int r = 0; r < 8; ++r) {
        int gm = m0 + wm * 32 + i * 16 + rowoff + r;
        int gn = n0 + wn * 64 + j * 16 + coloff;
        size_t idx = (size_t)gm * D_MODEL + gn;
        Y[idx] = acc[i][j][r] + bias[gn] + X[idx];
      }
}

// ---------------------------------------------------------------------------
// Launch
// ---------------------------------------------------------------------------
extern "C" void kernel_launch(void* const* d_in, const int* in_sizes, int n_in,
                              void* d_out, int out_size, void* d_ws, size_t ws_size,
                              hipStream_t stream) {
  const float* x      = (const float*)d_in[0];   // [4,2048,1024]
  const float* adj    = (const float*)d_in[1];   // [2048,2048]
  const float* w_qkv  = (const float*)d_in[2];   // [3072,1024]
  const float* w_out  = (const float*)d_in[3];   // [1024,1024]
  const float* b_out  = (const float*)d_in[4];   // [1024]
  const float* ln_g   = (const float*)d_in[5];
  const float* ln_b   = (const float*)d_in[6];
  (void)in_sizes; (void)n_in; (void)out_size; (void)ws_size;

  char* ws = (char*)d_ws;
  size_t off = 0;
  _Float16* Hh   = (_Float16*)(ws + off); off += (size_t)ROWS * D_MODEL * 2;        // 16 MB
  _Float16* Wq16 = (_Float16*)(ws + off); off += (size_t)3 * D_MODEL * D_MODEL * 2; //  6 MB
  _Float16* Wo16 = (_Float16*)(ws + off); off += (size_t)D_MODEL * D_MODEL * 2;     //  2 MB
  _Float16* Qb   = (_Float16*)(ws + off); off += (size_t)ROWS * D_MODEL * 2;        // 16 MB
  _Float16* Kb   = (_Float16*)(ws + off); off += (size_t)ROWS * D_MODEL * 2;        // 16 MB
  _Float16* Vt   = (_Float16*)(ws + off); off += (size_t)ROWS * D_MODEL * 2;        // 16 MB
  _Float16* Ao   = (_Float16*)(ws + off); off += (size_t)ROWS * D_MODEL * 2;        // 16 MB
  unsigned* Mb   = (unsigned*)(ws + off); off += (size_t)N_SEQ * 64 * 4;            // 0.5 MB

  ln_kernel<<<ROWS, 256, 0, stream>>>(x, ln_g, ln_b, Hh);
  cast_f16_kernel<<<(3 * D_MODEL * D_MODEL) / 256, 256, 0, stream>>>(
      w_qkv, Wq16, 3 * D_MODEL * D_MODEL);
  cast_f16_kernel<<<(D_MODEL * D_MODEL) / 256, 256, 0, stream>>>(
      w_out, Wo16, D_MODEL * D_MODEL);
  maskpack_kernel<<<(N_SEQ * 64) / 256, 256, 0, stream>>>(adj, Mb);

  gemm_qkv_kernel<<<dim3(ROWS / 64, (3 * D_MODEL) / GEMM_TN), 256, 0, stream>>>(
      Hh, Wq16, Qb, Kb, Vt);

  attn_kernel<<<BATCH * N_HEADS * (N_SEQ / 64), 128, 0, stream>>>(
      Qb, Kb, Vt, Mb, Ao);

  gemm_out_kernel<<<dim3(ROWS / 64, D_MODEL / GEMM_TN), 256, 0, stream>>>(
      Ao, Wo16, b_out, x, (float*)d_out);
}
